// TinyFasterRcnn_64750926955095
// MI455X (gfx1250) — matrix operand order repaired and checked by reading, not compiled, not run on hardware
//
#include <hip/hip_runtime.h>
#include <hip/hip_bf16.h>
#include <math.h>
#include <stdint.h>

// ---------------- problem constants ----------------
#define MDIM   256      // proposals
#define KDIM   200704   // 1024 * 14 * 14
#define NDIM   1024     // fc width
#define CFEAT  1024
#define HFEAT  50
#define WFEAT  80
#define NCLS   80
#define NLOG   81
#define NBOX   320
#define KPRE   1000
#define KOUT   200
#define NCAND  (MDIM * NCLS)          // 20480
#define SCALE_CLAMP 4.1351666f        // log(1000/16)

// GEMM tiling
#define NSLICES  16                    // 1024 / 64
#define KCHUNKS  32
#define KPC      (KDIM / KCHUNKS)      // 6272
#define KIT      (KPC / 32)            // 196

typedef __attribute__((ext_vector_type(16))) __bf16 v16bf;
typedef __attribute__((ext_vector_type(8)))  float  v8f;
typedef __attribute__((ext_vector_type(4)))  unsigned int v4u;
typedef __attribute__((ext_vector_type(8)))  unsigned int v8u;

union FragU { v16bf v; uint4 q[2]; };

// ---------------- workspace layout (bytes) ----------------
#define WS_A    ((size_t)0)                                  // bf16 A [256][200704]  102,760,448 B
#define WS_XP   (WS_A  + (size_t)MDIM * KDIM * 2)            // fp32 partials [32][256][1024]
#define WS_X    (WS_XP + (size_t)KCHUNKS * MDIM * NDIM * 4)  // fp32 x [256][1024]
#define WS_SC   (WS_X  + (size_t)MDIM * NDIM * 4)            // fp32 scores [256][80]
#define WS_BX   (WS_SC + (size_t)MDIM * NCLS * 4)            // fp32 boxes  [20480][4]
#define WS_CS   (WS_BX + (size_t)NCAND * 4 * 4)              // fp32 cand scores [1000]
#define WS_CB   (WS_CS + (size_t)4096)                       // fp32 cand boxes  [1000][4]
#define WS_CC   (WS_CB + (size_t)16384)                      // int  cand class  [1000]

// =====================================================================
// 1) RoI-align producer: features[1,1024,50,80] + proposals[256,4]
//    -> A (bf16) [256, 200704], k = c*196 + py*14 + px
// =====================================================================
__global__ __launch_bounds__(256) void roi_align_kernel(
    const float* __restrict__ feat, const float* __restrict__ props,
    __bf16* __restrict__ A) {
  size_t idx = (size_t)blockIdx.x * 256 + threadIdx.x;
  if (idx >= (size_t)MDIM * KDIM) return;
  int m = (int)(idx / KDIM);
  int r = (int)(idx % KDIM);
  int c = r / 196;
  int b = r % 196;
  int py = b / 14, px = b % 14;

  const float* pr = props + m * 4;
  float x1 = pr[0] * 0.0625f, y1 = pr[1] * 0.0625f;
  float x2 = pr[2] * 0.0625f, y2 = pr[3] * 0.0625f;
  float rw = fmaxf(x2 - x1, 1.0f), rh = fmaxf(y2 - y1, 1.0f);
  float bw = rw * (1.0f / 14.0f), bh = rh * (1.0f / 14.0f);
  float X = x1 + ((float)px + 0.5f) * bw;
  float Y = y1 + ((float)py + 0.5f) * bh;
  bool valid = (Y >= -1.0f) && (Y <= (float)HFEAT) && (X >= -1.0f) && (X <= (float)WFEAT);
  float y = fminf(fmaxf(Y, 0.0f), (float)(HFEAT - 1));
  float x = fminf(fmaxf(X, 0.0f), (float)(WFEAT - 1));
  int y0 = (int)floorf(y), x0 = (int)floorf(x);
  int y1i = min(y0 + 1, HFEAT - 1), x1i = min(x0 + 1, WFEAT - 1);
  float ly = y - (float)y0, lx = x - (float)x0;
  float hy = 1.0f - ly, hx = 1.0f - lx;
  const float* f = feat + (size_t)c * (HFEAT * WFEAT);
  float v = hy * hx * f[y0 * WFEAT + x0] + hy * lx * f[y0 * WFEAT + x1i]
          + ly * hx * f[y1i * WFEAT + x0] + ly * lx * f[y1i * WFEAT + x1i];
  A[idx] = (__bf16)(valid ? v : 0.0f);
}

// =====================================================================
// 2) FC GEMM: Xp[kc] = A[256,Kc] (bf16) @ W[Kc,1024] (fp32->bf16)
//    512 WGs = 16 N-slices x 32 K-chunks, 256 threads (8 waves).
//    A slab staged by the Tensor Data Mover (one tensor_load_to_lds per
//    iteration, TENSORcnt-tracked), double buffered so TDM + weight
//    conversion overlap the v_wmma_f32_16x16x32_bf16 stream.
// =====================================================================
__global__ __launch_bounds__(256) void fc_gemm_kernel(
    const __bf16* __restrict__ A, const float* __restrict__ W,
    float* __restrict__ Xp) {
  __shared__ __align__(16) __bf16 As[2][MDIM * 32];   // [m][k] 2 x 16 KB
  __shared__ __align__(16) __bf16 Bs[2][64 * 32];     // [n][k] 2 x 4 KB

  const int bid = blockIdx.x;
  const int ns  = bid & (NSLICES - 1);
  const int kc  = bid >> 4;
  const int n0  = ns * 64;
  const long k0 = (long)kc * KPC;
  const int t    = threadIdx.x;
  const int lane = t & 31;
  const int wv   = t >> 5;

  v8f acc[8] = {};

  const int rp = t >> 4;         // K row-pair id (0..15) -> rows {2rp, 2rp+1}
  const int ng = (t & 15) * 4;   // N column group (4 floats)

  const unsigned long long Abase = (unsigned long long)(size_t)A;
  const unsigned ldsA0 = (unsigned)(size_t)(void*)&As[0][0];
  const unsigned ldsA1 = (unsigned)(size_t)(void*)&As[1][0];

  // ---- TDM descriptor group 1 (constant): 2D tile, 2-byte elems,
  //      tensor_dim0=200704 (stride 200704), tensor_dim1=256,
  //      tile_dim0=32, tile_dim1=256, no multicast/padding/iterate.
  v8u g1;
  g1[0] = 1u << 16;                                                   // data_size=2B
  g1[1] = ((unsigned)KDIM & 0xffffu) << 16;                           // tensor_dim0 lo
  g1[2] = (((unsigned)KDIM >> 16) & 0xffffu) | ((unsigned)MDIM << 16);// dim0 hi | tensor_dim1 lo
  g1[3] = 32u << 16;                                                  // tile_dim0=32
  g1[4] = (unsigned)MDIM;                                             // tile_dim1=256
  g1[5] = (unsigned)KDIM;                                             // tensor_dim0_stride lo
  g1[6] = 0u;
  g1[7] = 0u;

  auto issue_tdm = [&](long kk, unsigned lds) {
    unsigned long long ga = Abase + (unsigned long long)kk * 2ull;
    v4u g0;
    g0[0] = 1u;                                                  // count=1, user mode
    g0[1] = lds;                                                 // lds_addr
    g0[2] = (unsigned)ga;                                        // global_addr[31:0]
    g0[3] = ((unsigned)(ga >> 32) & 0x01ffffffu) | (2u << 30);   // addr[56:32] | type=2
    asm volatile("tensor_load_to_lds %0, %1" :: "s"(g0), "s"(g1) : "memory");
  };

  auto load_w = [&](long kk, int buf) {
    const float* gw = W + (kk + (long)(2 * rp)) * (long)NDIM + (n0 + ng);
    float4 w0 = *(const float4*)gw;          // K row 2rp
    float4 w1 = *(const float4*)(gw + NDIM); // K row 2rp+1
    __builtin_prefetch(gw + 32 * NDIM, 0, 1);  // global_prefetch_b8 on weight stream
    union { __bf16 h[2]; unsigned u; } p0, p1, p2, p3;
    p0.h[0] = (__bf16)w0.x; p0.h[1] = (__bf16)w1.x;
    p1.h[0] = (__bf16)w0.y; p1.h[1] = (__bf16)w1.y;
    p2.h[0] = (__bf16)w0.z; p2.h[1] = (__bf16)w1.z;
    p3.h[0] = (__bf16)w0.w; p3.h[1] = (__bf16)w1.w;
    unsigned* bs = (unsigned*)&Bs[buf][0];   // dword index = n*16 + rp
    bs[(ng + 0) * 16 + rp] = p0.u;
    bs[(ng + 1) * 16 + rp] = p1.u;
    bs[(ng + 2) * 16 + rp] = p2.u;
    bs[(ng + 3) * 16 + rp] = p3.u;
  };

  // ---- prologue: stage tile 0
  if (wv == 0) issue_tdm(k0, ldsA0);
  load_w(k0, 0);
  if (wv == 0) __builtin_amdgcn_s_wait_tensorcnt(0);
  __syncthreads();

  const int l15 = lane & 15;
  const int koA = (lane < 16) ? 0 : 16;  // A: K {0-7,16-23} / {8-15,24-31}
  const int koB = (lane < 16) ? 0 : 16;  // B: K {0-7,16-23} / {8-15,24-31}

  for (int it = 0; it < KIT; ++it) {
    const int cb = it & 1, nb = cb ^ 1;

    // ---- stage next tile while computing this one
    if (it + 1 < KIT) {
      const long kk1 = k0 + (long)(it + 1) * 32;
      if (wv == 0) issue_tdm(kk1, nb ? ldsA1 : ldsA0);
      load_w(kk1, nb);
    }

    // ---- fragments + WMMA on current buffers
    const char* Ab = (const char*)&As[cb][0];
    const char* Bb = (const char*)&Bs[cb][0];

    FragU af[2];
#pragma unroll
    for (int mt = 0; mt < 2; ++mt) {
      const char* p = Ab + (((wv * 2 + mt) * 16 + l15) * 64) + koA;
      af[mt].q[0] = *(const uint4*)p;
      af[mt].q[1] = *(const uint4*)(p + 32);
    }
#pragma unroll
    for (int nt = 0; nt < 4; ++nt) {
      FragU bfr;
      const char* p = Bb + ((nt * 16 + l15) * 64) + koB;
      bfr.q[0] = *(const uint4*)p;
      bfr.q[1] = *(const uint4*)(p + 32);
#pragma unroll
      for (int mt = 0; mt < 2; ++mt) {
        acc[nt * 2 + mt] = __builtin_amdgcn_wmma_f32_16x16x32_bf16(
            false, af[mt].v, false, bfr.v, (short)0, acc[nt * 2 + mt],
            false, false);
      }
    }

    if (wv == 0) __builtin_amdgcn_s_wait_tensorcnt(0);  // drain TDM(it+1)
    __syncthreads();
  }

  // ---- epilogue: C/D layout -> partial buffer (deterministic split-K)
  float* Xmy = Xp + (size_t)kc * (MDIM * NDIM);
  const int mhalf = (lane < 16) ? 0 : 8;
#pragma unroll
  for (int mt = 0; mt < 2; ++mt) {
#pragma unroll
    for (int nt = 0; nt < 4; ++nt) {
      v8f a = acc[nt * 2 + mt];
      int mbase = (wv * 2 + mt) * 16 + mhalf;
      int nn = n0 + nt * 16 + l15;
#pragma unroll
      for (int r = 0; r < 8; ++r)
        Xmy[(size_t)(mbase + r) * NDIM + nn] = a[r];
    }
  }
}

// =====================================================================
// 3) Reduce split-K partials + bias -> X [256,1024]
// =====================================================================
__global__ __launch_bounds__(256) void reduce_x_kernel(
    const float* __restrict__ Xp, const float* __restrict__ b_fc,
    float* __restrict__ X) {
  int i = blockIdx.x * 256 + threadIdx.x;
  if (i >= MDIM * NDIM) return;
  float s = b_fc[i & (NDIM - 1)];
  for (int p = 0; p < KCHUNKS; ++p) s += Xp[(size_t)p * (MDIM * NDIM) + i];
  X[i] = s;
}

// =====================================================================
// 4) Heads: logits/softmax/scores + deltas/apply/clip -> boxes
//    one block per roi, 512 threads
// =====================================================================
__global__ __launch_bounds__(512) void heads_kernel(
    const float* __restrict__ X, const float* __restrict__ props,
    const float* __restrict__ w_cls, const float* __restrict__ b_cls,
    const float* __restrict__ w_box, const float* __restrict__ b_box,
    float* __restrict__ scores_g, float* __restrict__ boxes_g) {
  __shared__ float xs[NDIM];
  __shared__ float lg[NLOG];
  __shared__ float dl[NBOX];
  __shared__ float red[2];
  const int m = blockIdx.x, t = threadIdx.x;
  xs[t]       = X[m * NDIM + t];
  xs[t + 512] = X[m * NDIM + t + 512];
  __syncthreads();

  if (t < NBOX) {
    float a = b_box[t];
    for (int k = 0; k < NDIM; ++k) a = fmaf(xs[k], w_box[k * NBOX + t], a);
    dl[t] = a;
  } else if (t < NBOX + NLOG) {
    int c = t - NBOX;
    float a = b_cls[c];
    for (int k = 0; k < NDIM; ++k) a = fmaf(xs[k], w_cls[k * NLOG + c], a);
    lg[c] = a;
  }
  __syncthreads();
  if (t == 0) {
    float mx = lg[0];
    for (int c = 1; c < NLOG; ++c) mx = fmaxf(mx, lg[c]);
    float s = 0.0f;
    for (int c = 0; c < NLOG; ++c) s += expf(lg[c] - mx);
    red[0] = mx;
    red[1] = 1.0f / s;
  }
  __syncthreads();
  if (t < NCLS) {
    scores_g[m * NCLS + t] = expf(lg[t + 1] - red[0]) * red[1];
    const float* pr = props + m * 4;
    float bw = pr[2] - pr[0], bh = pr[3] - pr[1];
    float cx = pr[0] + 0.5f * bw, cy = pr[1] + 0.5f * bh;
    float dx = dl[4 * t + 0] * 0.1f, dy = dl[4 * t + 1] * 0.1f;
    float dw = fminf(dl[4 * t + 2] * 0.2f, SCALE_CLAMP);
    float dh = fminf(dl[4 * t + 3] * 0.2f, SCALE_CLAMP);
    float pcx = dx * bw + cx, pcy = dy * bh + cy;
    float pw = expf(dw) * bw, ph = expf(dh) * bh;
    float x1 = fminf(fmaxf(pcx - 0.5f * pw, 0.0f), 1280.0f);
    float y1 = fminf(fmaxf(pcy - 0.5f * ph, 0.0f), 800.0f);
    float x2 = fminf(fmaxf(pcx + 0.5f * pw, 0.0f), 1280.0f);
    float y2 = fminf(fmaxf(pcy + 0.5f * ph, 0.0f), 800.0f);
    float* ob = boxes_g + (size_t)(m * NCLS + t) * 4;
    ob[0] = x1; ob[1] = y1; ob[2] = x2; ob[3] = y2;
  }
}

// =====================================================================
// 5) Rank-based stable top-1000 of masked scores (matches lax.top_k order)
// =====================================================================
__global__ __launch_bounds__(256) void topk_kernel(
    const float* __restrict__ scores_g, const float* __restrict__ boxes_g,
    float* __restrict__ cand_s, float* __restrict__ cand_box,
    int* __restrict__ cand_cls) {
  int i = blockIdx.x * 256 + threadIdx.x;
  if (i >= NCAND) return;
  float si = scores_g[i];
  float mi = si > 0.5f ? si : -1.0f;
  int rank = 0;
  for (int j = 0; j < NCAND; ++j) {
    float sj = scores_g[j];
    float mj = sj > 0.5f ? sj : -1.0f;
    rank += (mj > mi) || (mj == mi && j < i);
  }
  if (rank < KPRE) {
    cand_s[rank] = mi;
    cand_cls[rank] = i % NCLS;
    const float* b = boxes_g + (size_t)i * 4;
    float* o = cand_box + (size_t)rank * 4;
    o[0] = b[0]; o[1] = b[1]; o[2] = b[2]; o[3] = b[3];
  }
}

// =====================================================================
// 6) Class-offset NMS (sequential scan) + compaction to 200 outputs
//    d_out layout: boxes[200*4] | scores[200] | classes[200]  (floats)
// =====================================================================
__global__ __launch_bounds__(1024) void nms_kernel(
    const float* __restrict__ cand_s, const float* __restrict__ cand_box,
    const int* __restrict__ cand_cls, float* __restrict__ out) {
  __shared__ float cb[KPRE * 4];
  __shared__ float ca[KPRE];
  __shared__ unsigned char kept[KPRE];
  __shared__ short slot[KPRE];
  const int t = threadIdx.x;

  for (int i = t; i < KPRE; i += 1024) {
    float off = (float)cand_cls[i] * 1281.0f;
    float x1 = cand_box[i * 4 + 0] + off, y1 = cand_box[i * 4 + 1] + off;
    float x2 = cand_box[i * 4 + 2] + off, y2 = cand_box[i * 4 + 3] + off;
    cb[i * 4 + 0] = x1; cb[i * 4 + 1] = y1;
    cb[i * 4 + 2] = x2; cb[i * 4 + 3] = y2;
    ca[i] = fmaxf(x2 - x1, 0.0f) * fmaxf(y2 - y1, 0.0f);
    kept[i] = cand_s[i] > 0.5f;
  }
  for (int i = 0; i < KPRE; ++i) {
    __syncthreads();
    if (kept[i]) {
      float ax1 = cb[i * 4], ay1 = cb[i * 4 + 1];
      float ax2 = cb[i * 4 + 2], ay2 = cb[i * 4 + 3], ai = ca[i];
      for (int j = i + 1 + t; j < KPRE; j += 1024) {
        float ix1 = fmaxf(ax1, cb[j * 4]), iy1 = fmaxf(ay1, cb[j * 4 + 1]);
        float ix2 = fminf(ax2, cb[j * 4 + 2]), iy2 = fminf(ay2, cb[j * 4 + 3]);
        float iw = fmaxf(ix2 - ix1, 0.0f), ih = fmaxf(iy2 - iy1, 0.0f);
        float inter = iw * ih;
        float iou = inter / fmaxf(ai + ca[j] - inter, 1e-9f);
        if (iou > 0.7f) kept[j] = 0;
      }
    }
  }
  __syncthreads();
  if (t == 0) {
    int r = 0;
    for (int i = 0; i < KPRE; ++i) slot[i] = kept[i] ? (short)(r++) : (short)KOUT;
  }
  __syncthreads();
  for (int i = t; i < 1200; i += 1024) out[i] = (i < 1000) ? 0.0f : -1.0f;
  __syncthreads();
  for (int i = t; i < KPRE; i += 1024) {
    int s = slot[i];
    if (s < KOUT) {
      out[s * 4 + 0] = cand_box[i * 4 + 0];
      out[s * 4 + 1] = cand_box[i * 4 + 1];
      out[s * 4 + 2] = cand_box[i * 4 + 2];
      out[s * 4 + 3] = cand_box[i * 4 + 3];
      out[800 + s] = cand_s[i];
      out[1000 + s] = (float)cand_cls[i];
    }
  }
}

// =====================================================================
extern "C" void kernel_launch(void* const* d_in, const int* in_sizes, int n_in,
                              void* d_out, int out_size, void* d_ws, size_t ws_size,
                              hipStream_t stream) {
  const float* features  = (const float*)d_in[0];
  const float* proposals = (const float*)d_in[1];
  const float* w_fc      = (const float*)d_in[2];
  const float* b_fc      = (const float*)d_in[3];
  const float* w_cls     = (const float*)d_in[4];
  const float* b_cls     = (const float*)d_in[5];
  const float* w_box     = (const float*)d_in[6];
  const float* b_box     = (const float*)d_in[7];

  char* ws = (char*)d_ws;
  __bf16* A       = (__bf16*)(ws + WS_A);
  float*  Xp      = (float*)(ws + WS_XP);
  float*  X       = (float*)(ws + WS_X);
  float*  scoresg = (float*)(ws + WS_SC);
  float*  boxesg  = (float*)(ws + WS_BX);
  float*  cand_s  = (float*)(ws + WS_CS);
  float*  cand_b  = (float*)(ws + WS_CB);
  int*    cand_c  = (int*)(ws + WS_CC);
  float*  out     = (float*)d_out;

  roi_align_kernel<<<(int)(((size_t)MDIM * KDIM + 255) / 256), 256, 0, stream>>>(
      features, proposals, A);
  fc_gemm_kernel<<<NSLICES * KCHUNKS, 256, 0, stream>>>(A, w_fc, Xp);
  reduce_x_kernel<<<(MDIM * NDIM + 255) / 256, 256, 0, stream>>>(Xp, b_fc, X);
  heads_kernel<<<MDIM, 512, 0, stream>>>(X, proposals, w_cls, b_cls, w_box, b_box,
                                         scoresg, boxesg);
  topk_kernel<<<(NCAND + 255) / 256, 256, 0, stream>>>(scoresg, boxesg,
                                                       cand_s, cand_b, cand_c);
  nms_kernel<<<1, 1024, 0, stream>>>(cand_s, cand_b, cand_c, out);
}
